// ACGI_32195074850822
// MI455X (gfx1250) — matrix operand order repaired and checked by reading, not compiled, not run on hardware
//
#include <hip/hip_runtime.h>
#include <math.h>

typedef unsigned short u16;
typedef __attribute__((ext_vector_type(16))) __bf16 v16bf;
typedef __attribute__((ext_vector_type(8)))  float  v8f;

static const int Dd = 512;
static const int Ll = 4;
static const int Bz = 8;
static const int Ns = 1024;
static const int Mr = Bz * Ns;        // 8192
static const int WM = Dd * Dd;        // 262144 elems per weight matrix
static const int MD = Mr * Dd;        // 4194304 elems per activation
static const float SC = 0.044194173824159216f;  // 512^-0.5

#define TM 128
#define TN 128
#define KSTEP 32
#define LPITCH 40   // halves per LDS row: 32 data + 8 pad = 80 bytes (16B aligned)

__device__ __forceinline__ u16 f2bf(float x) {
  union { float f; unsigned u; } c; c.f = x;
  unsigned r = c.u + 0x7FFFu + ((c.u >> 16) & 1u);   // round-to-nearest-even
  return (u16)(r >> 16);
}

// gfx1250 async copy: global -> LDS, 16 bytes per lane, tracked by ASYNCcnt.
__device__ __forceinline__ void async_ld16(unsigned lds_off, const void* gptr) {
  asm volatile("global_load_async_to_lds_b128 %0, %1, off"
               :: "v"(lds_off), "v"((unsigned long long)(size_t)gptr)
               : "memory");
}
__device__ __forceinline__ void wait_async0() {
  asm volatile("s_wait_asynccnt 0x0" ::: "memory");
}

// ---------------------------------------------------------------------------
// Batched bf16 WMMA GEMM:  C[b] = scale * (A[b] @ Bt[b]^T) + bias (+ C[b])
//   A  row-major [M,K] bf16,  Bt row-major [N,K] bf16 (i.e. B pre-transposed)
//   Cf f32 out (optional, accum optional), Ch bf16 out (optional)
// Block: 256 threads = 8 wave32, tile 128x128, K-step 32, LDS double-buffered
// via global_load_async_to_lds_b128. Each wave: 32x64 sub-tile = 2x4 fragments
// of v_wmma_f32_16x16x32_bf16.
// ---------------------------------------------------------------------------
__global__ __launch_bounds__(256) void k_gemm(
    const u16* __restrict__ A, long sAb,
    const u16* __restrict__ Bt, long sBb,
    const float* __restrict__ bias, float scale,
    float* __restrict__ Cf, u16* __restrict__ Ch, long sCb,
    int accum, int Ndim, int Kdim)
{
  __shared__ __align__(16) u16 As[2][TM * LPITCH];
  __shared__ __align__(16) u16 Bs[2][TN * LPITCH];

  const int bz = blockIdx.z;
  A  += (long)bz * sAb;
  Bt += (long)bz * sBb;
  const long cb = (long)bz * sCb;
  const int m0 = blockIdx.y * TM;
  const int n0 = blockIdx.x * TN;
  const int t    = threadIdx.x;
  const int lane = t & 31;
  const int wave = t >> 5;
  const int mbase = (wave >> 1) * 32;   // 4 wave-rows
  const int nbase = (wave & 1) * 64;    // 2 wave-cols
  const int r  = lane & 15;
  const int hi = lane >> 4;

  const unsigned aB[2] = { (unsigned)(size_t)&As[0][0], (unsigned)(size_t)&As[1][0] };
  const unsigned bB[2] = { (unsigned)(size_t)&Bs[0][0], (unsigned)(size_t)&Bs[1][0] };

  // loader: 128 rows x 4 chunks(16B) per tile = 512 chunks; 256 threads -> 2 each
  const int c0row = t >> 2;
  const int c0off = (t & 3) * 8;
  const int c1row = (t + 256) >> 2;
  const int c1off = ((t + 256) & 3) * 8;
  const unsigned l0 = (unsigned)((c0row * LPITCH + c0off) * 2);
  const unsigned l1 = (unsigned)((c1row * LPITCH + c1off) * 2);

  const u16* gA = A  + (long)m0 * Kdim;
  const u16* gB = Bt + (long)n0 * Kdim;

  v8f acc[2][4] = {};

  const int nk = Kdim / KSTEP;
  // prologue: stage K-tile 0 into buffer 0
  async_ld16(aB[0] + l0, gA + (long)c0row * Kdim + c0off);
  async_ld16(bB[0] + l0, gB + (long)c0row * Kdim + c0off);
  async_ld16(aB[0] + l1, gA + (long)c1row * Kdim + c1off);
  async_ld16(bB[0] + l1, gB + (long)c1row * Kdim + c1off);

  for (int it = 0; it < nk; ++it) {
    wait_async0();        // my async loads for buf(it&1) have landed
    __syncthreads();      // everyone's landed; prior reads of other buf are done
    if (it + 1 < nk) {    // overlap next tile's DMA with this tile's WMMAs
      int kk = (it + 1) * KSTEP;
      int nb = (it + 1) & 1;
      async_ld16(aB[nb] + l0, gA + (long)c0row * Kdim + kk + c0off);
      async_ld16(bB[nb] + l0, gB + (long)c0row * Kdim + kk + c0off);
      async_ld16(aB[nb] + l1, gA + (long)c1row * Kdim + kk + c1off);
      async_ld16(bB[nb] + l1, gB + (long)c1row * Kdim + kk + c1off);
    }
    const u16* Asc = &As[it & 1][0];
    const u16* Bsc = &Bs[it & 1][0];

    // A fragment (16x32): lane<16 -> M=lane, K=0..7 & 16..23; lane>=16 -> K=8..15 & 24..31
    union Frag { uint4 q[2]; v16bf v; } a[2], b[4];
#pragma unroll
    for (int i = 0; i < 2; ++i) {
      const u16* ar = Asc + (mbase + i * 16 + r) * LPITCH + hi * 8;
      a[i].q[0] = *(const uint4*)(ar);
      a[i].q[1] = *(const uint4*)(ar + 16);
    }
    // B fragment (32x16): N=lane&15, lane<16 -> K=0..7 & 16..23; lane>=16 -> K=8..15 & 24..31
#pragma unroll
    for (int j = 0; j < 4; ++j) {
      const u16* br = Bsc + (nbase + j * 16 + r) * LPITCH + hi * 8;
      b[j].q[0] = *(const uint4*)(br);
      b[j].q[1] = *(const uint4*)(br + 16);
    }
#pragma unroll
    for (int i = 0; i < 2; ++i)
#pragma unroll
      for (int j = 0; j < 4; ++j)
        acc[i][j] = __builtin_amdgcn_wmma_f32_16x16x32_bf16(
            false, a[i].v, false, b[j].v, (short)0, acc[i][j], false, false);
  }

  // Epilogue. C layout: lane col = lane&15 (per 16-col group), VGPR e -> row e (+8 hi lanes).
  const int lr = hi * 8;
#pragma unroll
  for (int j = 0; j < 4; ++j) {
    int gcol = n0 + nbase + j * 16 + r;
    float bv = bias ? bias[gcol] : 0.0f;
#pragma unroll
    for (int i = 0; i < 2; ++i) {
      int grow = m0 + mbase + i * 16 + lr;
#pragma unroll
      for (int e = 0; e < 8; ++e) {
        long idx = cb + (long)(grow + e) * Ndim + gcol;
        float v = acc[i][j][e] * scale + bv;
        if (accum) v += Cf[idx];
        if (Cf) Cf[idx] = v;
        if (Ch) Ch[idx] = f2bf(v);
      }
    }
  }
}

// ---------------------------------------------------------------------------
// f32 -> bf16 convert
__global__ void k_cvt(const float* __restrict__ in, u16* __restrict__ out, int n) {
  int i = blockIdx.x * 256 + threadIdx.x;
  if (i < n) out[i] = f2bf(in[i]);
}

// Weight convert + transpose: f32 W[K,N] -> bf16 Wt[N,K]; grid.z = matrix index
__global__ void k_wt(const float* __restrict__ W, u16* __restrict__ Wt, int K, int N) {
  __shared__ u16 tile[32][33];
  const long mo = (long)blockIdx.z * (long)K * N;
  int n0 = blockIdx.x * 32, k0 = blockIdx.y * 32;
  for (int i = threadIdx.y; i < 32; i += 8)
    tile[i][threadIdx.x] = f2bf(W[mo + (long)(k0 + i) * N + n0 + threadIdx.x]);
  __syncthreads();
  for (int i = threadIdx.y; i < 32; i += 8)
    Wt[mo + (long)(n0 + i) * K + k0 + threadIdx.x] = tile[threadIdx.x][i];
}

// Batched bf16 transpose: in[b][R][C] -> out[b][C][R]
__global__ void k_trh(const u16* __restrict__ in, u16* __restrict__ out, int R, int C) {
  __shared__ u16 tile[32][33];
  const long base = (long)blockIdx.z * (long)R * C;
  int c0 = blockIdx.x * 32, r0 = blockIdx.y * 32;
  for (int i = threadIdx.y; i < 32; i += 8)
    tile[i][threadIdx.x] = in[base + (long)(r0 + i) * C + c0 + threadIdx.x];
  __syncthreads();
  for (int i = threadIdx.y; i < 32; i += 8)
    out[base + (long)(c0 + i) * R + r0 + threadIdx.x] = tile[threadIdx.x][i];
}

// Row softmax: f32 S row -> bf16 P row; one block (256 thr) per row
__global__ void k_softmax(const float* __restrict__ S, u16* __restrict__ P, int len) {
  __shared__ float red[256];
  long row = blockIdx.x;
  const float* x = S + row * (long)len;
  u16* y = P + row * (long)len;
  float m = -INFINITY;
  for (int i = threadIdx.x; i < len; i += 256) m = fmaxf(m, x[i]);
  red[threadIdx.x] = m; __syncthreads();
  for (int s = 128; s > 0; s >>= 1) {
    if (threadIdx.x < s) red[threadIdx.x] = fmaxf(red[threadIdx.x], red[threadIdx.x + s]);
    __syncthreads();
  }
  m = red[0]; __syncthreads();
  float sum = 0.f;
  for (int i = threadIdx.x; i < len; i += 256) sum += __expf(x[i] - m);
  red[threadIdx.x] = sum; __syncthreads();
  for (int s = 128; s > 0; s >>= 1) {
    if (threadIdx.x < s) red[threadIdx.x] += red[threadIdx.x + s];
    __syncthreads();
  }
  float inv = 1.0f / red[0];
  for (int i = threadIdx.x; i < len; i += 256) y[i] = f2bf(__expf(x[i] - m) * inv);
}

// out = L2-normalize over axis=1 (sequence) of (a + maybeLeaky(b)); shapes (B,Nr,Dd)
__global__ void k_addnorm(const float* __restrict__ a, const float* __restrict__ b,
                          int leaky_b, float* __restrict__ outF, u16* __restrict__ outH,
                          int Nr, int Ddim) {
  int batch = blockIdx.y;
  int d = blockIdx.x * 128 + threadIdx.x;
  long base = (long)batch * Nr * Ddim + d;
  float ss = 0.f;
  for (int n = 0; n < Nr; ++n) {
    float v = a[base + (long)n * Ddim];
    float w = b[base + (long)n * Ddim];
    if (leaky_b) w = w >= 0.f ? w : 0.01f * w;
    v += w; ss += v * v;
  }
  float inv = 1.0f / fmaxf(sqrtf(ss), 1e-12f);
  for (int n = 0; n < Nr; ++n) {
    float v = a[base + (long)n * Ddim];
    float w = b[base + (long)n * Ddim];
    if (leaky_b) w = w >= 0.f ? w : 0.01f * w;
    v = (v + w) * inv;
    outF[base + (long)n * Ddim] = v;
    if (outH) outH[base + (long)n * Ddim] = f2bf(v);
  }
}

// LeakyReLU, dual output (f32 + bf16)
__global__ void k_leaky(const float* __restrict__ in, float* __restrict__ outF,
                        u16* __restrict__ outH, int n) {
  int i = blockIdx.x * 256 + threadIdx.x;
  if (i >= n) return;
  float v = in[i];
  v = v >= 0.f ? v : 0.01f * v;
  outF[i] = v;
  outH[i] = f2bf(v);
}

// out = a + b, dual output (f32 + bf16)
__global__ void k_add_cvt(const float* __restrict__ a, const float* __restrict__ b,
                          float* __restrict__ outF, u16* __restrict__ outH, int n) {
  int i = blockIdx.x * 256 + threadIdx.x;
  if (i >= n) return;
  float v = a[i] + b[i];
  outF[i] = v;
  outH[i] = f2bf(v);
}

// ---------------------------------------------------------------------------
// Host orchestration
// ---------------------------------------------------------------------------
struct Ws {
  u16 *pTH, *pPH, *pRH, *pPS, *trA, *trB, *Sh, *accb, *resH, *h1b;
  float *Sf, *acc1, *acc2, *uB, *resF, *h2;
};

static void launch_gemm(hipStream_t st, const void* A, long sA, const void* Bt, long sB,
                        const float* bias, float scale, float* Cf, u16* Ch, long sC,
                        int accum, int M, int N, int K, int batch) {
  dim3 grid(N / TN, M / TM, batch);
  k_gemm<<<grid, 256, 0, st>>>((const u16*)A, sA, (const u16*)Bt, sB,
                               bias, scale, Cf, Ch, sC, accum, N, K);
}

// One AGI block: softmax self-attention x4 (avg), residual + seq-norm, FFN+leaky, norm.
static void run_agi(hipStream_t st, const Ws& w, const float* xF, const u16* xH,
                    const u16* thW, const float* thB, const u16* phW, const float* phB,
                    const u16* rhW, const float* rhB, const u16* intW, const float* intB,
                    const u16* f1W, const float* f1B, const u16* f2W, const float* f2B,
                    float* outF, u16* outH) {
  for (int i = 0; i < Ll; ++i) {
    launch_gemm(st, xH, 0, thW + (size_t)i * WM, 0, thB + i * Dd, 1.f, nullptr, w.pTH, 0, 0, Mr, Dd, Dd, 1);
    launch_gemm(st, xH, 0, phW + (size_t)i * WM, 0, phB + i * Dd, 1.f, nullptr, w.pPH, 0, 0, Mr, Dd, Dd, 1);
    launch_gemm(st, xH, 0, rhW + (size_t)i * WM, 0, rhB + i * Dd, 1.f, nullptr, w.pRH, 0, 0, Mr, Dd, Dd, 1);
    // scores: S = SC * th @ ph^T  (ph rows already along K=D)
    launch_gemm(st, w.pTH, (long)Ns * Dd, w.pPH, (long)Ns * Dd, nullptr, SC,
                w.Sf, nullptr, (long)Ns * Ns, 0, Ns, Ns, Dd, Bz);
    k_softmax<<<Bz * Ns, 256, 0, st>>>(w.Sf, w.Sh, Ns);
    k_trh<<<dim3(Dd / 32, Ns / 32, Bz), dim3(32, 8), 0, st>>>(w.pRH, w.trA, Ns, Dd);
    // acc += P @ rh
    launch_gemm(st, w.Sh, (long)Ns * Ns, w.trA, (long)Dd * Ns, nullptr, 1.f,
                w.acc1, nullptr, (long)Ns * Dd, i > 0, Ns, Dd, Ns, Bz);
  }
  k_cvt<<<MD / 256, 256, 0, st>>>(w.acc1, w.accb, MD);
  launch_gemm(st, w.accb, 0, intW, 0, intB, 1.0f / Ll, w.uB, nullptr, 0, 0, Mr, Dd, Dd, 1);
  k_addnorm<<<dim3(Dd / 128, Bz), 128, 0, st>>>(xF, w.uB, 0, w.resF, w.resH, Ns, Dd);
  launch_gemm(st, w.resH, 0, f1W, 0, f1B, 1.f, nullptr, w.h1b, 0, 0, Mr, Dd, Dd, 1);
  launch_gemm(st, w.h1b, 0, f2W, 0, f2B, 1.f, w.h2, nullptr, 0, 0, Mr, Dd, Dd, 1);
  k_addnorm<<<dim3(Dd / 128, Bz), 128, 0, st>>>(w.resF, w.h2, 1, outF, outH, Ns, Dd);
}

extern "C" void kernel_launch(void* const* d_in, const int* in_sizes, int n_in,
                              void* d_out, int out_size, void* d_ws, size_t ws_size,
                              hipStream_t stream) {
  (void)in_sizes; (void)n_in; (void)out_size; (void)ws_size;
  char* base = (char*)d_ws;
  size_t off = 0;
  auto take = [&](size_t bytes) -> char* {
    off = (off + 255) & ~(size_t)255;
    char* q = base + off;
    off += bytes;
    return q;
  };

  // bf16 transposed weights
  u16* w_cth  = (u16*)take((size_t)Ll * WM * 2);
  u16* w_cph  = (u16*)take((size_t)Ll * WM * 2);
  u16* w_crh  = (u16*)take((size_t)Ll * WM * 2);
  u16* w_cps  = (u16*)take((size_t)Ll * WM * 2);
  u16* w_cint = (u16*)take((size_t)WM * 2);
  u16* w_cf1  = (u16*)take((size_t)WM * 2);
  u16* w_cf2  = (u16*)take((size_t)WM * 2);
  u16* w_a1th = (u16*)take((size_t)Ll * WM * 2);
  u16* w_a1ph = (u16*)take((size_t)Ll * WM * 2);
  u16* w_a1rh = (u16*)take((size_t)Ll * WM * 2);
  u16* w_a1in = (u16*)take((size_t)WM * 2);
  u16* w_a1f1 = (u16*)take((size_t)WM * 2);
  u16* w_a1f2 = (u16*)take((size_t)WM * 2);
  u16* w_a2th = (u16*)take((size_t)Ll * WM * 2);
  u16* w_a2ph = (u16*)take((size_t)Ll * WM * 2);
  u16* w_a2rh = (u16*)take((size_t)Ll * WM * 2);
  u16* w_a2in = (u16*)take((size_t)WM * 2);
  u16* w_a2f1 = (u16*)take((size_t)WM * 2);
  u16* w_a2f2 = (u16*)take((size_t)WM * 2);

  // activations
  u16* x1b = (u16*)take((size_t)MD * 2);
  u16* x2b = (u16*)take((size_t)MD * 2);
  Ws w;
  w.pTH  = (u16*)take((size_t)MD * 2);
  w.pPH  = (u16*)take((size_t)MD * 2);
  w.pRH  = (u16*)take((size_t)MD * 2);
  w.pPS  = (u16*)take((size_t)MD * 2);
  w.trA  = (u16*)take((size_t)MD * 2);
  w.trB  = (u16*)take((size_t)MD * 2);
  w.Sh   = (u16*)take((size_t)Bz * Ns * Ns * 2);
  w.accb = (u16*)take((size_t)MD * 2);
  w.resH = (u16*)take((size_t)MD * 2);
  w.h1b  = (u16*)take((size_t)MD * 2);
  w.Sf   = (float*)take((size_t)Bz * Ns * Ns * 4);
  w.acc1 = (float*)take((size_t)MD * 4);
  w.acc2 = (float*)take((size_t)MD * 4);
  w.uB   = (float*)take((size_t)MD * 4);
  w.resF = (float*)take((size_t)MD * 4);
  w.h2   = (float*)take((size_t)MD * 4);
  float* feed1F = (float*)take((size_t)MD * 4);
  u16*   feed1H = (u16*)take((size_t)MD * 2);
  float* feed2F = (float*)take((size_t)MD * 4);
  u16*   feed2H = (u16*)take((size_t)MD * 2);
  float* s1F = (float*)take((size_t)MD * 4);
  float* s2F = (float*)take((size_t)MD * 4);
  float* xsF = (float*)take((size_t)MD * 4);
  u16*   xsH = (u16*)take((size_t)MD * 2);

  const float* in1 = (const float*)d_in[0];
  const float* in2 = (const float*)d_in[1];

  // Convert inputs + all weights (transposed) to bf16
  k_cvt<<<MD / 256, 256, 0, stream>>>(in1, x1b, MD);
  k_cvt<<<MD / 256, 256, 0, stream>>>(in2, x2b, MD);
  dim3 wg(Dd / 32, Dd / 32, Ll), wg1(Dd / 32, Dd / 32, 1), wb(32, 8);
  k_wt<<<wg,  wb, 0, stream>>>((const float*)d_in[2],  w_cth,  Dd, Dd);
  k_wt<<<wg,  wb, 0, stream>>>((const float*)d_in[4],  w_cph,  Dd, Dd);
  k_wt<<<wg,  wb, 0, stream>>>((const float*)d_in[6],  w_crh,  Dd, Dd);
  k_wt<<<wg,  wb, 0, stream>>>((const float*)d_in[8],  w_cps,  Dd, Dd);
  k_wt<<<wg1, wb, 0, stream>>>((const float*)d_in[10], w_cint, Dd, Dd);
  k_wt<<<wg1, wb, 0, stream>>>((const float*)d_in[12], w_cf1,  Dd, Dd);
  k_wt<<<wg1, wb, 0, stream>>>((const float*)d_in[14], w_cf2,  Dd, Dd);
  k_wt<<<wg,  wb, 0, stream>>>((const float*)d_in[16], w_a1th, Dd, Dd);
  k_wt<<<wg,  wb, 0, stream>>>((const float*)d_in[18], w_a1ph, Dd, Dd);
  k_wt<<<wg,  wb, 0, stream>>>((const float*)d_in[20], w_a1rh, Dd, Dd);
  k_wt<<<wg1, wb, 0, stream>>>((const float*)d_in[22], w_a1in, Dd, Dd);
  k_wt<<<wg1, wb, 0, stream>>>((const float*)d_in[24], w_a1f1, Dd, Dd);
  k_wt<<<wg1, wb, 0, stream>>>((const float*)d_in[26], w_a1f2, Dd, Dd);
  k_wt<<<wg,  wb, 0, stream>>>((const float*)d_in[28], w_a2th, Dd, Dd);
  k_wt<<<wg,  wb, 0, stream>>>((const float*)d_in[30], w_a2ph, Dd, Dd);
  k_wt<<<wg,  wb, 0, stream>>>((const float*)d_in[32], w_a2rh, Dd, Dd);
  k_wt<<<wg1, wb, 0, stream>>>((const float*)d_in[34], w_a2in, Dd, Dd);
  k_wt<<<wg1, wb, 0, stream>>>((const float*)d_in[36], w_a2f1, Dd, Dd);
  k_wt<<<wg1, wb, 0, stream>>>((const float*)d_in[38], w_a2f2, Dd, Dd);

  // ---- Cross block (no softmax) ----
  for (int i = 0; i < Ll; ++i) {
    launch_gemm(stream, x1b, 0, w_cth + (size_t)i * WM, 0, (const float*)d_in[3] + i * Dd, 1.f, nullptr, w.pTH, 0, 0, Mr, Dd, Dd, 1);
    launch_gemm(stream, x1b, 0, w_crh + (size_t)i * WM, 0, (const float*)d_in[7] + i * Dd, 1.f, nullptr, w.pRH, 0, 0, Mr, Dd, Dd, 1);
    launch_gemm(stream, x2b, 0, w_cph + (size_t)i * WM, 0, (const float*)d_in[5] + i * Dd, 1.f, nullptr, w.pPH, 0, 0, Mr, Dd, Dd, 1);
    launch_gemm(stream, x2b, 0, w_cps + (size_t)i * WM, 0, (const float*)d_in[9] + i * Dd, 1.f, nullptr, w.pPS, 0, 0, Mr, Dd, Dd, 1);
    // aff (scaled, bf16, no softmax)
    launch_gemm(stream, w.pTH, (long)Ns * Dd, w.pPH, (long)Ns * Dd, nullptr, SC,
                nullptr, w.Sh, (long)Ns * Ns, 0, Ns, Ns, Dd, Bz);
    k_trh<<<dim3(Dd / 32, Ns / 32, Bz), dim3(32, 8), 0, stream>>>(w.pRH, w.trA, Ns, Dd);
    k_trh<<<dim3(Dd / 32, Ns / 32, Bz), dim3(32, 8), 0, stream>>>(w.pPS, w.trB, Ns, Dd);
    launch_gemm(stream, w.Sh, (long)Ns * Ns, w.trA, (long)Dd * Ns, nullptr, 1.f,
                w.acc1, nullptr, (long)Ns * Dd, i > 0, Ns, Dd, Ns, Bz);
    launch_gemm(stream, w.Sh, (long)Ns * Ns, w.trB, (long)Dd * Ns, nullptr, 1.f,
                w.acc2, nullptr, (long)Ns * Dd, i > 0, Ns, Dd, Ns, Bz);
  }
  // stream 1 tail: res1 -> feed1
  k_cvt<<<MD / 256, 256, 0, stream>>>(w.acc1, w.accb, MD);
  launch_gemm(stream, w.accb, 0, w_cint, 0, (const float*)d_in[11], 1.0f / Ll, w.uB, nullptr, 0, 0, Mr, Dd, Dd, 1);
  k_addnorm<<<dim3(Dd / 128, Bz), 128, 0, stream>>>(in1, w.uB, 0, w.resF, w.resH, Ns, Dd);
  launch_gemm(stream, w.resH, 0, w_cf1, 0, (const float*)d_in[13], 1.f, nullptr, w.h1b, 0, 0, Mr, Dd, Dd, 1);
  launch_gemm(stream, w.h1b, 0, w_cf2, 0, (const float*)d_in[15], 1.f, w.h2, nullptr, 0, 0, Mr, Dd, Dd, 1);
  k_leaky<<<MD / 256, 256, 0, stream>>>(w.h2, feed1F, feed1H, MD);
  // stream 2 tail: res2 -> feed2
  k_cvt<<<MD / 256, 256, 0, stream>>>(w.acc2, w.accb, MD);
  launch_gemm(stream, w.accb, 0, w_cint, 0, (const float*)d_in[11], 1.0f / Ll, w.uB, nullptr, 0, 0, Mr, Dd, Dd, 1);
  k_addnorm<<<dim3(Dd / 128, Bz), 128, 0, stream>>>(in2, w.uB, 0, w.resF, w.resH, Ns, Dd);
  launch_gemm(stream, w.resH, 0, w_cf1, 0, (const float*)d_in[13], 1.f, nullptr, w.h1b, 0, 0, Mr, Dd, Dd, 1);
  launch_gemm(stream, w.h1b, 0, w_cf2, 0, (const float*)d_in[15], 1.f, w.h2, nullptr, 0, 0, Mr, Dd, Dd, 1);
  k_leaky<<<MD / 256, 256, 0, stream>>>(w.h2, feed2F, feed2H, MD);

  // ---- AGI blocks: self1 = AGI1(feed1), self2 = AGI1(feed2), out = AGI2(self1+self2) ----
  run_agi(stream, w, feed1F, feed1H,
          w_a1th, (const float*)d_in[17], w_a1ph, (const float*)d_in[19],
          w_a1rh, (const float*)d_in[21], w_a1in, (const float*)d_in[23],
          w_a1f1, (const float*)d_in[25], w_a1f2, (const float*)d_in[27],
          s1F, nullptr);
  run_agi(stream, w, feed2F, feed2H,
          w_a1th, (const float*)d_in[17], w_a1ph, (const float*)d_in[19],
          w_a1rh, (const float*)d_in[21], w_a1in, (const float*)d_in[23],
          w_a1f1, (const float*)d_in[25], w_a1f2, (const float*)d_in[27],
          s2F, nullptr);
  k_add_cvt<<<MD / 256, 256, 0, stream>>>(s1F, s2F, xsF, xsH, MD);
  run_agi(stream, w, xsF, xsH,
          w_a2th, (const float*)d_in[29], w_a2ph, (const float*)d_in[31],
          w_a2rh, (const float*)d_in[33], w_a2in, (const float*)d_in[35],
          w_a2f1, (const float*)d_in[37], w_a2f2, (const float*)d_in[39],
          (float*)d_out, nullptr);
}
